// EqvMSPFeedForward_51994874085877
// MI455X (gfx1250) — matrix operand order repaired and checked by reading, not compiled, not run on hardware
//
#include <hip/hip_runtime.h>
#include <hip/hip_bf16.h>
#include <math.h>

// ---------------------------------------------------------------------------
// EqvMSPFeedForward for MI455X (gfx1250, wave32, WMMA).
// Factorized:  out[z,a,o] = (1/(sqrt(H)*sqrt(N))) * sum_{b,h} h2[z,a,b,h] * G[z,b,h,o]
// where        G[z,b,h,o] = sum_i w3[h, o*C+i] * x[z,b,i]
// Three f16 WMMA GEMMs (f32 accum).  A row-major, B stored K-contiguous per
// column (Bt[n][k]) so both fragments load straight from global with b128
// loads -- no LDS, no barriers in the GEMM.  All fragment loads for a K-step
// are issued before the 4-WMMA chain so waits are partial (latency overlap).
// Split-K x32 (deterministic reduce) for the small per-batch GEMM3.
// H padded 100->128 so every GEMM dim is a multiple of 64.
// ---------------------------------------------------------------------------

typedef __attribute__((ext_vector_type(16))) _Float16 v16h;
typedef __attribute__((ext_vector_type(8)))  float    v8f;
typedef __attribute__((ext_vector_type(4)))  unsigned u32x4;

#define HPAD 128      // H=100 padded to 128
#define NPAIR 32768   // 8*64*64 pairs
#define HO 8192       // HPAD*64 = (h,o) flattened
#define ZSLAB 524288  // 64*8192 per-z slab of h2 / Gt, in halves
#define NSPLIT 32     // split-K factor for GEMM3

// ---------------- packing kernels (f32 -> padded f16) ----------------------

// w2t[n][k] = w2[k][n]  (zero pad), shape [128][128]  (B of GEMM1, K-contiguous)
__global__ void pack_w2t(const float* __restrict__ w2, _Float16* __restrict__ w2t) {
    int idx = blockIdx.x * 256 + threadIdx.x;
    if (idx >= HPAD * HPAD) return;
    int n = idx >> 7, k = idx & 127;
    float v = (k < 100 && n < 100) ? w2[k * 100 + n] : 0.0f;
    w2t[idx] = (_Float16)v;
}

// w3bt[h*64+o][i] = w3[h][o*64+i]  (zero for h>=100), shape [8192][64]
__global__ void pack_w3bt(const float* __restrict__ w3, _Float16* __restrict__ w3bt) {
    int idx = blockIdx.x * 256 + threadIdx.x;
    if (idx >= HO * 64) return;
    int row = idx >> 6, i = idx & 63;
    int h = row >> 6, o = row & 63;
    float v = (h < 100) ? w3[h * 4096 + o * 64 + i] : 0.0f;
    w3bt[idx] = (_Float16)v;
}

// xp[z*64+b][i] = concat(input1,input2)  -> f16 [512][64]
__global__ void pack_x(const float* __restrict__ in1, const float* __restrict__ in2,
                       _Float16* __restrict__ xp) {
    int idx = blockIdx.x * 256 + threadIdx.x;
    if (idx >= 512 * 64) return;
    int row = idx >> 6, i = idx & 63;
    int z = row >> 6, b = row & 63;
    float v = (b < 32) ? in1[(z * 32 + b) * 64 + i] : in2[(z * 32 + (b - 32)) * 64 + i];
    xp[idx] = (_Float16)v;
}

// ---------------- radial basis + first MLP layer ---------------------------
__global__ __launch_bounds__(128) void radial_h1(
    const float* __restrict__ xyz1, const float* __restrict__ xyz2,
    const float* __restrict__ w1, _Float16* __restrict__ h1) {
    __shared__ float basis[10];
    int p = blockIdx.x;
    int n = threadIdx.x;
    int z = p >> 12, a = (p >> 6) & 63, b = p & 63;
    if (n < 10) {
        const float* pa = (a < 32) ? &xyz1[(z * 32 + a) * 3] : &xyz2[(z * 32 + (a - 32)) * 3];
        const float* pb = (b < 32) ? &xyz1[(z * 32 + b) * 3] : &xyz2[(z * 32 + (b - 32)) * 3];
        float dx = pa[0] - pb[0], dy = pa[1] - pb[1], dz = pa[2] - pb[2];
        float r = sqrtf(dx * dx + dy * dy + dz * dz + 1e-12f);
        const float step = 10.0f / 9.0f;
        const float hpi = 1.5707963267948966f;
        float u = (r - (float)n * step) / step * hpi;
        float c = cosf(u);
        basis[n] = (fabsf(u) < hpi) ? c * c : 0.0f;
    }
    __syncthreads();
    float acc = 0.0f;
    if (n < 100) {
#pragma unroll
        for (int j = 0; j < 10; ++j) acc += basis[j] * w1[j * 100 + n];
        acc *= 0.31622776601683794f;          // 1/sqrt(10)
        acc = acc / (1.0f + expf(-acc));      // silu
    }
    h1[(size_t)p * HPAD + n] = (_Float16)acc; // pads (n>=100) written as 0
}

// ---------------- generic WMMA GEMM, direct-register, no LDS ---------------
// C = epilogue(scale * A@B).  A f16 [M,K] row-major lda.  Bt f16 [N][K]
// row-major ldbt.  K % 32 == 0.  128 threads = 4 waves; wave w computes rows
// 16w..16w+15 of a 64x64 tile (1 A frag x 4 B frags = 4 WMMA per K-step).
// EPI: 0 = f32 linear, 2 = silu -> f16 linear, 3 = f16 scatter to
//      Gt[z][o][b*128+h] (row=(z,b), col=(h,o)).
template <int EPI>
__global__ __launch_bounds__(128) void wmma_gemm(
    const _Float16* __restrict__ A, const _Float16* __restrict__ Bt,
    void* __restrict__ Cout, int lda, int ldbt, int ldc,
    long long strideA, long long strideB, long long strideC, float scale,
    int nSplit, int kChunk, long long splitStride) {
    int t = threadIdx.x;
    int wave = t >> 5, lane = t & 31;
    int l15 = lane & 15, hi = lane >> 4;
    int z = blockIdx.z / nSplit;
    int split = blockIdx.z % nSplit;
    int kStart = split * kChunk, kEnd = kStart + kChunk;
    const _Float16* Arow = A + (long long)z * strideA +
                           (size_t)(blockIdx.x * 64 + wave * 16 + l15) * lda;
    const _Float16* Brow0 = Bt + (long long)z * strideB +
                            (size_t)(blockIdx.y * 64 + l15) * ldbt;
    v8f acc[4] = {};
    union frag { v16h v; u32x4 q[2]; };
    for (int kt = kStart; kt < kEnd; kt += 32) {
        // issue ALL fragment loads first (one clause), then the WMMA chain,
        // so the compiler can use partial loadcnt waits and overlap.
        frag afr, bfr[4];
        afr.q[0] = *(const u32x4*)(Arow + kt + hi * 8);
        afr.q[1] = *(const u32x4*)(Arow + kt + hi * 8 + 16);
#pragma unroll
        for (int s = 0; s < 4; ++s) {
            const _Float16* br = Brow0 + (size_t)s * 16 * ldbt + kt + hi * 8;
            bfr[s].q[0] = *(const u32x4*)(br);
            bfr[s].q[1] = *(const u32x4*)(br + 16);
        }
        // prefetch next K-tile (global_prefetch_b8)
        int ktn = kt + 32;
        if (ktn < kEnd) {
            __builtin_prefetch((const void*)(Arow + ktn + hi * 16), 0, 3);
            __builtin_prefetch((const void*)(Brow0 + (size_t)wave * 16 * ldbt + ktn + hi * 16), 0, 3);
        }
#pragma unroll
        for (int s = 0; s < 4; ++s)
            acc[s] = __builtin_amdgcn_wmma_f32_16x16x32_f16(
                false, afr.v, false, bfr[s].v, (short)0, acc[s], false, false);
    }
    // C layout: reg r -> m = wave*16 + r + hi*8, n = s*16 + l15
    int mrow = blockIdx.x * 64 + wave * 16 + hi * 8;
#pragma unroll
    for (int s = 0; s < 4; ++s) {
        int ncol = blockIdx.y * 64 + s * 16 + l15;
#pragma unroll
        for (int r = 0; r < 8; ++r) {
            float v = acc[s][r] * scale;
            if (EPI == 2) v = v / (1.0f + expf(-v));
            if (EPI == 3) {
                int row = mrow + r, col = ncol;
                int zz = row >> 6, b = row & 63, h = col >> 6, o = col & 63;
                size_t off = (size_t)(zz * 64 + o) * (size_t)HO + (size_t)b * HPAD + h;
                ((_Float16*)Cout)[off] = (_Float16)v;
            } else {
                size_t off = (size_t)split * splitStride + (size_t)z * strideC +
                             (size_t)(mrow + r) * ldc + ncol;
                if (EPI == 0) ((float*)Cout)[off] = v;
                else          ((_Float16*)Cout)[off] = (_Float16)v;
            }
        }
    }
}

// ---------------- split-K reduction (deterministic fixed-order sum) --------
__global__ void reduce_split(const float* __restrict__ part, float* __restrict__ xout) {
    int idx = blockIdx.x * 256 + threadIdx.x;
    if (idx >= 32768) return;
    float s = 0.0f;
    for (int sp = 0; sp < NSPLIT; ++sp) s += part[(size_t)sp * 32768 + idx];
    xout[idx] = s;
}

// ---------------- head: mask/abs/sum, layernorm, fc3+leaky, fc2, sigmoid ---
__global__ __launch_bounds__(64) void finalize(
    const float* __restrict__ xout, const int* __restrict__ mask,
    const float* __restrict__ fc3, const float* __restrict__ fc2,
    float* __restrict__ out) {
    __shared__ float s[64], red[64];
    int z = blockIdx.x, o = threadIdx.x;
    float acc = 0.0f;
    for (int a = 0; a < 64; ++a)
        if (mask[z * 64 + a] != 0)
            acc += fabsf(xout[((size_t)z * 64 + a) * 64 + o]);
    red[o] = acc; __syncthreads();
    for (int st = 32; st > 0; st >>= 1) { if (o < st) red[o] += red[o + st]; __syncthreads(); }
    float mean = red[0] * (1.0f / 64.0f);
    __syncthreads();
    float d = acc - mean;
    red[o] = d * d; __syncthreads();
    for (int st = 32; st > 0; st >>= 1) { if (o < st) red[o] += red[o + st]; __syncthreads(); }
    float stdv = sqrtf(red[0] * (1.0f / 63.0f));   // unbiased (ddof=1)
    __syncthreads();
    s[o] = d / (stdv + 1e-6f);
    __syncthreads();
    float h = 0.0f;
    for (int i = 0; i < 64; ++i) h += s[i] * fc3[i * 64 + o];
    h *= 0.125f;                                   // 1/sqrt(64)
    h = (h > 0.0f) ? h : 0.01f * h;                // leaky_relu
    red[o] = h * fc2[o] * 0.125f; __syncthreads();
    for (int st = 32; st > 0; st >>= 1) { if (o < st) red[o] += red[o + st]; __syncthreads(); }
    if (o == 0) out[z] = 1.0f / (1.0f + expf(-red[0]));
}

// ---------------------------------------------------------------------------

extern "C" void kernel_launch(void* const* d_in, const int* in_sizes, int n_in,
                              void* d_out, int out_size, void* d_ws, size_t ws_size,
                              hipStream_t stream) {
    const float* input1 = (const float*)d_in[0];   // [8,32,64]
    const float* input2 = (const float*)d_in[1];   // [8,32,64]
    const float* xyz1   = (const float*)d_in[2];   // [8,32,3]
    const float* xyz2   = (const float*)d_in[3];   // [8,32,3]
    const int*   mask   = (const int*)  d_in[4];   // [8,64]
    const float* w1     = (const float*)d_in[5];   // [10,100]
    const float* w2     = (const float*)d_in[6];   // [100,100]
    const float* w3     = (const float*)d_in[7];   // [100,4096]
    const float* fc3    = (const float*)d_in[8];   // [64,64]
    const float* fc2    = (const float*)d_in[9];   // [64,1]
    float* out = (float*)d_out;                    // [8]

    char* ws = (char*)d_ws;
    _Float16* h1b  = (_Float16*)(ws);                  // 32768*128 f16 = 8,388,608 B
    _Float16* w2t  = (_Float16*)(ws + 8388608);        // 128*128 f16   =    32,768 B
    _Float16* h2b  = (_Float16*)(ws + 8421376);        // 32768*128 f16 = 8,388,608 B
    _Float16* w3bt = (_Float16*)(ws + 16809984);       // 8192*64 f16   = 1,048,576 B
    _Float16* xp   = (_Float16*)(ws + 17858560);       // 512*64 f16    =    65,536 B
    _Float16* Gt   = (_Float16*)(ws + 17924096);       // 8*64*8192 f16 = 8,388,608 B
    float*    part = (float*)   (ws + 26312704);       // 32*8*4096 f32 = 4,194,304 B
    float*    xout = (float*)   (ws + 30507008);       // 8*64*64 f32   =   131,072 B
    (void)ws_size; (void)n_in; (void)in_sizes; (void)out_size;

    pack_w2t<<<(HPAD * HPAD + 255) / 256, 256, 0, stream>>>(w2, w2t);
    pack_w3bt<<<(HO * 64 + 255) / 256, 256, 0, stream>>>(w3, w3bt);
    pack_x <<<(512 * 64 + 255) / 256, 256, 0, stream>>>(input1, input2, xp);
    radial_h1<<<NPAIR, 128, 0, stream>>>(xyz1, xyz2, w1, h1b);

    // GEMM1: h2 = silu(0.1 * h1 @ w2)    [32768 x 128] @ [128 x 128] -> f16 linear
    wmma_gemm<2><<<dim3(NPAIR / 64, HPAD / 64, 1), 128, 0, stream>>>(
        h1b, w2t, h2b, HPAD, HPAD, HPAD, 0, 0, 0, 0.1f, 1, HPAD, 0);

    // GEMM2: G = x @ W3^T                [512 x 64] @ [64 x 8192] -> f16 scatter Gt
    wmma_gemm<3><<<dim3(512 / 64, HO / 64, 1), 128, 0, stream>>>(
        xp, w3bt, Gt, 64, 64, 0, 0, 0, 0, 1.0f, 1, 64, 0);

    // GEMM3 (per z, split-K 32): part = (1/80) * h2[z] @ G[z]  [64x8192]@[8192x64]
    wmma_gemm<0><<<dim3(1, 1, 8 * NSPLIT), 128, 0, stream>>>(
        h2b, Gt, part, HO, HO, 64,
        (long long)ZSLAB, (long long)ZSLAB, 4096LL, 0.0125f, NSPLIT, HO / NSPLIT, 32768LL);

    reduce_split<<<128, 256, 0, stream>>>(part, xout);
    finalize<<<8, 64, 0, stream>>>(xout, mask, fc3, fc2, out);
}
